// AdvancedEventDKG_71201967833691
// MI455X (gfx1250) — compile-verified, matrix operand order checked by reading, not run on hardware
//
#include <hip/hip_runtime.h>
#include <cstddef>
#include <cstdint>

// ---------------------------------------------------------------------------
// MI455X (gfx1250) implementation of AdvancedEventDKG forward.
// Dense GEMMs use v_wmma_f32_16x16x32_bf16 (bf16 in, f32 accumulate).
// Edge softmax/aggregation is streaming + atomics (memory-bound part:
// ~4 GB total @ 23.3 TB/s HBM dominates; GEMMs ~25 GFLOP are cheap on WMMA).
// ---------------------------------------------------------------------------

#define DEVINL __device__ __forceinline__

typedef __attribute__((ext_vector_type(16))) __bf16        v16bf;
typedef __attribute__((ext_vector_type(8)))  float         v8f;
typedef __attribute__((ext_vector_type(4)))  unsigned int  v4u;

union BF16Frag { v16bf v; v4u q[2]; unsigned short h[16]; };

static constexpr int Nn = 20000;   // nodes (multiple of 16)
static constexpr int Ee = 250000;  // edges

// float -> bf16, round to nearest even
DEVINL unsigned short f2bf(float f) {
  unsigned int u = __float_as_uint(f);
  u = u + 0x7FFFu + ((u >> 16) & 1u);
  return (unsigned short)(u >> 16);
}
// order-preserving float <-> uint for atomicMax-based segment max
DEVINL unsigned fenc(float f) {
  unsigned u = __float_as_uint(f);
  return (u & 0x80000000u) ? ~u : (u | 0x80000000u);
}
DEVINL float fdec(unsigned k) {
  unsigned u = (k & 0x80000000u) ? (k & 0x7FFFFFFFu) : ~k;
  return __uint_as_float(u);
}
DEVINL float wredsum(float v) {
#pragma unroll
  for (int off = 16; off > 0; off >>= 1) v += __shfl_xor(v, off, 32);
  return v;
}
DEVINL float sigmoidf_(float x) { return 1.0f / (1.0f + __expf(-x)); }

// ---------------------------------------------------------------------------
// Utility fills
// ---------------------------------------------------------------------------
__global__ void fill_f32_kernel(float* p, float v, int n) {
  int i = blockIdx.x * blockDim.x + threadIdx.x;
  if (i < n) p[i] = v;
}
__global__ void fill_u32_kernel(unsigned* p, unsigned v, int n) {
  int i = blockIdx.x * blockDim.x + threadIdx.x;
  if (i < n) p[i] = v;
}

// ---------------------------------------------------------------------------
// Event encoder: e = relu(LN(relu(LN(f@w1+b1))@w2+b2)) + mean_k ett[id]*int
// One block of 128 threads; LN reductions via shared-memory tree.
// ---------------------------------------------------------------------------
__global__ void event_encoder_kernel(
    const float* __restrict__ ev_feat, const int* __restrict__ etype,
    const float* __restrict__ eint, const float* __restrict__ ett,
    const float* __restrict__ w1, const float* __restrict__ b1,
    const float* __restrict__ g1, const float* __restrict__ bt1,
    const float* __restrict__ w2, const float* __restrict__ b2,
    const float* __restrict__ g2, const float* __restrict__ bt2,
    float* __restrict__ e_out) {
  __shared__ float val[128];
  __shared__ float red[128];
  int t = threadIdx.x;

  // fc1: [64] @ [64,128]
  float a = b1[t];
  for (int i = 0; i < 64; ++i) a += ev_feat[i] * w1[i * 128 + t];
  red[t] = a; __syncthreads();
  for (int s = 64; s > 0; s >>= 1) { if (t < s) red[t] += red[t + s]; __syncthreads(); }
  float mu = red[0] * (1.0f / 128.0f); __syncthreads();
  float dm = a - mu;
  red[t] = dm * dm; __syncthreads();
  for (int s = 64; s > 0; s >>= 1) { if (t < s) red[t] += red[t + s]; __syncthreads(); }
  float var = red[0] * (1.0f / 128.0f); __syncthreads();
  float y = fmaxf(dm * rsqrtf(var + 1e-5f) * g1[t] + bt1[t], 0.0f);
  val[t] = y; __syncthreads();

  // fc2: [128] @ [128,128]
  float a2 = b2[t];
  for (int i = 0; i < 128; ++i) a2 += val[i] * w2[i * 128 + t];
  red[t] = a2; __syncthreads();
  for (int s = 64; s > 0; s >>= 1) { if (t < s) red[t] += red[t + s]; __syncthreads(); }
  float mu2 = red[0] * (1.0f / 128.0f); __syncthreads();
  float dm2 = a2 - mu2;
  red[t] = dm2 * dm2; __syncthreads();
  for (int s = 64; s > 0; s >>= 1) { if (t < s) red[t] += red[t + s]; __syncthreads(); }
  float var2 = red[0] * (1.0f / 128.0f);
  float y2 = fmaxf(dm2 * rsqrtf(var2 + 1e-5f) * g2[t] + bt2[t], 0.0f);

  float add = 0.0f;
  for (int k = 0; k < 4; ++k) add += ett[etype[k] * 128 + t] * eint[k];
  e_out[t] = y2 + add * 0.25f;
}

// ---------------------------------------------------------------------------
// x = concat(node_emb[node_ids], broadcast(e)) as bf16  [N, 256]
// ---------------------------------------------------------------------------
__global__ void build_x256_kernel(const int* __restrict__ node_ids,
                                  const float* __restrict__ emb,
                                  const float* __restrict__ e,
                                  unsigned short* __restrict__ xb, int total) {
  int i = blockIdx.x * blockDim.x + threadIdx.x;
  if (i >= total) return;
  int row = i >> 8, col = i & 255;
  float v = (col < 128) ? emb[(size_t)node_ids[row] * 128 + col] : e[col - 128];
  xb[i] = f2bf(v);
}

__global__ void f32_to_bf16_kernel(const float* __restrict__ in,
                                   unsigned short* __restrict__ out, int n) {
  int i = blockIdx.x * blockDim.x + threadIdx.x;
  if (i < n) out[i] = f2bf(in[i]);
}

// W[Kd][Nout] f32 row-major -> Wt[Nout][Kd] bf16 (column-major B for WMMA)
__global__ void convert_wT_kernel(const float* __restrict__ W,
                                  unsigned short* __restrict__ Wt, int Kd, int Nout) {
  int i = blockIdx.x * blockDim.x + threadIdx.x;
  if (i >= Kd * Nout) return;
  int k = i / Nout, n = i - k * Nout;
  Wt[(size_t)n * Kd + k] = f2bf(W[i]);
}

// ---------------------------------------------------------------------------
// WMMA GEMM: C[M,Nout] = A[M,Kd](bf16) @ B(bf16, stored as Bt[Nout,Kd]) + bias
// One wave computes a 16x64 C tile: 1 A-frag reused over 4 B-frags per K step.
// Fragment layouts per CDNA5 ISA 7.12.2 (16-bit A 16x32; C/D 32-bit 16x16).
// ---------------------------------------------------------------------------
__global__ void wmma_gemm_kernel(const unsigned short* __restrict__ A,
                                 const unsigned short* __restrict__ Bt,
                                 const float* __restrict__ bias,
                                 float* __restrict__ C,
                                 int M, int Kd, int Nout) {
  int gwave = (int)((blockIdx.x * blockDim.x + threadIdx.x) >> 5);
  int lane  = threadIdx.x & 31;
  int ncg   = Nout >> 6;                 // column groups of 64
  int total = (M >> 4) * ncg;
  if (gwave >= total) return;            // wave-uniform: EXEC stays all-ones
  int mtile = gwave / ncg;
  int cg    = gwave - mtile * ncg;
  int row0  = mtile << 4;
  int col0  = cg << 6;
  int r  = lane & 15;
  int hi = lane >> 4;

  const unsigned short* arow = A  + (size_t)(row0 + r) * Kd;
  const unsigned short* bc0  = Bt + (size_t)(col0 +  0 + r) * Kd;
  const unsigned short* bc1  = Bt + (size_t)(col0 + 16 + r) * Kd;
  const unsigned short* bc2  = Bt + (size_t)(col0 + 32 + r) * Kd;
  const unsigned short* bc3  = Bt + (size_t)(col0 + 48 + r) * Kd;

  v8f acc0 = {0.f,0.f,0.f,0.f,0.f,0.f,0.f,0.f};
  v8f acc1 = acc0, acc2 = acc0, acc3 = acc0;

  for (int k0 = 0; k0 < Kd; k0 += 32) {
    int o1 = k0 + 8 * hi;        // K = 8*hi .. 8*hi+7
    int o2 = k0 + 16 + 8 * hi;   // K = 16+8*hi .. +7
    BF16Frag fa, fb0, fb1, fb2, fb3;
    fa.q[0]  = *(const v4u*)(arow + o1); fa.q[1]  = *(const v4u*)(arow + o2);
    fb0.q[0] = *(const v4u*)(bc0 + o1);  fb0.q[1] = *(const v4u*)(bc0 + o2);
    fb1.q[0] = *(const v4u*)(bc1 + o1);  fb1.q[1] = *(const v4u*)(bc1 + o2);
    fb2.q[0] = *(const v4u*)(bc2 + o1);  fb2.q[1] = *(const v4u*)(bc2 + o2);
    fb3.q[0] = *(const v4u*)(bc3 + o1);  fb3.q[1] = *(const v4u*)(bc3 + o2);
    acc0 = __builtin_amdgcn_wmma_f32_16x16x32_bf16(false, fa.v, false, fb0.v, (short)0, acc0, false, false);
    acc1 = __builtin_amdgcn_wmma_f32_16x16x32_bf16(false, fa.v, false, fb1.v, (short)0, acc1, false, false);
    acc2 = __builtin_amdgcn_wmma_f32_16x16x32_bf16(false, fa.v, false, fb2.v, (short)0, acc2, false, false);
    acc3 = __builtin_amdgcn_wmma_f32_16x16x32_bf16(false, fa.v, false, fb3.v, (short)0, acc3, false, false);
  }

  // C/D layout: lane -> column n = r; VGPR rr -> row m = 8*hi + rr
  int n0 = col0 + r;
  float b0 = bias ? bias[n0]      : 0.0f;
  float b1 = bias ? bias[n0 + 16] : 0.0f;
  float b2 = bias ? bias[n0 + 32] : 0.0f;
  float b3 = bias ? bias[n0 + 48] : 0.0f;
#pragma unroll
  for (int rr = 0; rr < 8; ++rr) {
    int m = row0 + hi * 8 + rr;
    float* crow = C + (size_t)m * Nout;
    crow[n0]      = acc0[rr] + b0;
    crow[n0 + 16] = acc1[rr] + b1;
    crow[n0 + 32] = acc2[rr] + b2;
    crow[n0 + 48] = acc3[rr] + b3;
  }
}

// ---------------------------------------------------------------------------
// TransformerConv attention logits: alpha[e,h] = <q[dst],k[src]>_h / sqrt(128)
// One wave per edge; also seeds segment-max via encoded atomicMax.
// ---------------------------------------------------------------------------
__global__ void attn_logits_kernel(const float* __restrict__ q, const float* __restrict__ k,
                                   const int* __restrict__ src, const int* __restrict__ dst,
                                   int E, float* __restrict__ logits, unsigned* __restrict__ mEnc) {
  int gw   = (int)((blockIdx.x * blockDim.x + threadIdx.x) >> 5);
  int lane = threadIdx.x & 31;
  if (gw >= E) return;
  int s = src[gw], d = dst[gw];
  const float* qd = q + (size_t)d * 512;
  const float* ks = k + (size_t)s * 512;
  const float scale = 0.08838834764831845f;  // 1/sqrt(128)
#pragma unroll
  for (int h = 0; h < 4; ++h) {
    float p = 0.0f;
#pragma unroll
    for (int j = 0; j < 4; ++j) {
      int c = h * 128 + lane + 32 * j;
      p += qd[c] * ks[c];
    }
    p = wredsum(p) * scale;
    if (lane == 0) {
      logits[(size_t)gw * 4 + h] = p;
      atomicMax(&mEnc[d * 4 + h], fenc(p));
    }
  }
}

// GAT logits with implicit self loops: e < E -> (src,dst); e >= E -> (n,n)
__global__ void gat_logits_kernel(const float* __restrict__ as_, const float* __restrict__ ad_,
                                  const int* __restrict__ src, const int* __restrict__ dst,
                                  int E, int E2, float* __restrict__ logits,
                                  unsigned* __restrict__ mEnc) {
  int i = blockIdx.x * blockDim.x + threadIdx.x;
  if (i >= E2 * 4) return;
  int e = i >> 2, h = i & 3;
  int s, d;
  if (e < E) { s = src[e]; d = dst[e]; } else { s = e - E; d = s; }
  float l = as_[s * 4 + h] + ad_[d * 4 + h];
  l = (l > 0.0f) ? l : 0.2f * l;            // leaky_relu(0.2)
  logits[i] = l;
  atomicMax(&mEnc[d * 4 + h], fenc(l));
}

// exp(l - segmax) in place + segment sum via atomicAdd
__global__ void softmax_exp_kernel(float* __restrict__ logits, const int* __restrict__ dst,
                                   int E, int E2, const unsigned* __restrict__ mEnc,
                                   float* __restrict__ ssum) {
  int i = blockIdx.x * blockDim.x + threadIdx.x;
  if (i >= E2 * 4) return;
  int e = i >> 2, h = i & 3;
  int d = (e < E) ? dst[e] : (e - E);
  float m  = fdec(mEnc[d * 4 + h]);
  float ex = __expf(logits[i] - m);
  logits[i] = ex;
  atomicAdd(&ssum[d * 4 + h], ex);
}

// Weighted scatter with head-mean folded in: out[d,c] += scale*sum_h feat[s,h,c]*w[e,h]
__global__ void aggregate_kernel(const float* __restrict__ feat, const float* __restrict__ wexp,
                                 const float* __restrict__ ssum, const int* __restrict__ src,
                                 const int* __restrict__ dst, int E, int E2, float scale,
                                 float* __restrict__ out) {
  int gw   = (int)((blockIdx.x * blockDim.x + threadIdx.x) >> 5);
  int lane = threadIdx.x & 31;
  if (gw >= E2) return;
  int s, d;
  if (gw < E) { s = src[gw]; d = dst[gw]; } else { s = gw - E; d = s; }
  const float* we = wexp + (size_t)gw * 4;
  float w0 = we[0] / (ssum[d * 4 + 0] + 1e-16f) * scale;
  float w1 = we[1] / (ssum[d * 4 + 1] + 1e-16f) * scale;
  float w2 = we[2] / (ssum[d * 4 + 2] + 1e-16f) * scale;
  float w3 = we[3] / (ssum[d * 4 + 3] + 1e-16f) * scale;
  const float* fs = feat + (size_t)s * 512;
  float* od = out + (size_t)d * 128;
#pragma unroll
  for (int j = 0; j < 4; ++j) {
    int c = lane + 32 * j;
    float a = w0 * fs[c] + w1 * fs[128 + c] + w2 * fs[256 + c] + w3 * fs[384 + c];
    atomicAdd(&od[c], a);
  }
}

// GAT attention coefficients: a_s[n,h] = <xh[n,h,:], asrc[h,:]>, same for a_d
__global__ void gat_coef_kernel(const float* __restrict__ xh, const float* __restrict__ as_w,
                                const float* __restrict__ ad_w, float* __restrict__ as_o,
                                float* __restrict__ ad_o, int N) {
  int gw   = (int)((blockIdx.x * blockDim.x + threadIdx.x) >> 5);
  int lane = threadIdx.x & 31;
  if (gw >= N) return;
  const float* x = xh + (size_t)gw * 512;
#pragma unroll
  for (int h = 0; h < 4; ++h) {
    float ps = 0.0f, pd = 0.0f;
#pragma unroll
    for (int j = 0; j < 4; ++j) {
      int c = h * 128 + lane + 32 * j;
      float v = x[c];
      ps += v * as_w[c];
      pd += v * ad_w[c];
    }
    ps = wredsum(ps); pd = wredsum(pd);
    if (lane == 0) { as_o[gw * 4 + h] = ps; ad_o[gw * 4 + h] = pd; }
  }
}

__global__ void gat_final_kernel(const float* __restrict__ acc, const float* __restrict__ b,
                                 float* __restrict__ x, int total) {
  int i = blockIdx.x * blockDim.x + threadIdx.x;
  if (i >= total) return;
  x[i] = fmaxf(acc[i] + b[i & 127], 0.0f);
}

__global__ void bias_sum_kernel(const float* a, const float* b, float* o, int n) {
  int i = blockIdx.x * blockDim.x + threadIdx.x;
  if (i < n) o[i] = a[i] + b[i];
}

// LSTM cell, h0 = c0 = 0 (so f-gate term vanishes). Torch gate order i,f,g,o.
__global__ void lstm_final_kernel(const float* __restrict__ gates, float* __restrict__ out, int N) {
  int i = blockIdx.x * blockDim.x + threadIdx.x;
  if (i >= N * 128) return;
  int n = i >> 7, c = i & 127;
  const float* g = gates + (size_t)n * 512;
  float gi = g[c], gg = g[256 + c], go = g[384 + c];
  float cc = sigmoidf_(gi) * tanhf(gg);
  float hh = sigmoidf_(go) * tanhf(cc);
  out[i] = hh;                       // h
  out[(size_t)N * 128 + i] = cc;     // c
}

// ---------------------------------------------------------------------------
// Host launch
// ---------------------------------------------------------------------------
static inline int cdiv(long long a, long long b) { return (int)((a + b - 1) / b); }

extern "C" void kernel_launch(void* const* d_in, const int* in_sizes, int n_in,
                              void* d_out, int out_size, void* d_ws, size_t ws_size,
                              hipStream_t stream) {
  (void)in_sizes; (void)n_in; (void)out_size;
  const int N = Nn, E = Ee, E2 = Ee + Nn;

  const int*   node_ids = (const int*)d_in[0];
  const int*   etype    = (const int*)d_in[1];
  const int*   src      = (const int*)d_in[2];
  const int*   dst      = (const int*)d_in[3];
  const float* ev_feat  = (const float*)d_in[4];
  const float* eint     = (const float*)d_in[5];
  const float* emb      = (const float*)d_in[6];
  const float* ett      = (const float*)d_in[7];
  const float* ev_w1 = (const float*)d_in[8],  *ev_b1 = (const float*)d_in[9];
  const float* ev_g1 = (const float*)d_in[10], *ev_bt1 = (const float*)d_in[11];
  const float* ev_w2 = (const float*)d_in[12], *ev_b2 = (const float*)d_in[13];
  const float* ev_g2 = (const float*)d_in[14], *ev_bt2 = (const float*)d_in[15];
  const float* wq = (const float*)d_in[16], *bq = (const float*)d_in[17];
  const float* wk = (const float*)d_in[18], *bk = (const float*)d_in[19];
  const float* wv = (const float*)d_in[20], *bv = (const float*)d_in[21];
  const float* wskip = (const float*)d_in[22], *bskip = (const float*)d_in[23];
  const float* g_w[2]  = {(const float*)d_in[24], (const float*)d_in[28]};
  const float* g_as[2] = {(const float*)d_in[25], (const float*)d_in[29]};
  const float* g_ad[2] = {(const float*)d_in[26], (const float*)d_in[30]};
  const float* g_b[2]  = {(const float*)d_in[27], (const float*)d_in[31]};
  const float* wih = (const float*)d_in[32];
  const float* bih = (const float*)d_in[34], *bhh = (const float*)d_in[35];

  // ---- workspace arena (256B aligned) ----
  char* base = (char*)d_ws;
  size_t off = 0;
  auto alloc = [&](size_t bytes) -> char* {
    off = (off + 255) & ~(size_t)255;
    char* p = base + off;
    off += bytes;
    return p;
  };
  float*          e_buf  = (float*)alloc(128 * 4);
  float*          bs     = (float*)alloc(512 * 4);
  unsigned short* x256b  = (unsigned short*)alloc((size_t)N * 256 * 2);
  unsigned short* wT     = (unsigned short*)alloc((size_t)512 * 256 * 2);
  float*          qbuf   = (float*)alloc((size_t)N * 512 * 4);  // also xh, gates
  float*          kbuf   = (float*)alloc((size_t)N * 512 * 4);
  float*          vbuf   = (float*)alloc((size_t)N * 512 * 4);
  float*          logits = (float*)alloc((size_t)E2 * 4 * 4);
  unsigned*       mEnc   = (unsigned*)alloc((size_t)N * 4 * 4);
  float*          ssum   = (float*)alloc((size_t)N * 4 * 4);
  float*          asb    = (float*)alloc((size_t)N * 4 * 4);
  float*          adb    = (float*)alloc((size_t)N * 4 * 4);
  float*          xcur   = (float*)alloc((size_t)N * 128 * 4);
  float*          accb   = (float*)alloc((size_t)N * 128 * 4);
  unsigned short* xb16   = (unsigned short*)alloc((size_t)N * 128 * 2);
  if (off > ws_size) return;  // insufficient scratch: bail deterministically

  auto gemm = [&](const unsigned short* A, const unsigned short* Bt, const float* bias,
                  float* C, int M, int Kd, int Nout) {
    int waves = (M / 16) * (Nout / 64);
    wmma_gemm_kernel<<<cdiv(waves, 4), 128, 0, stream>>>(A, Bt, bias, C, M, Kd, Nout);
  };
  auto convw = [&](const float* W, int Kd, int Nout) {
    convert_wT_kernel<<<cdiv((long long)Kd * Nout, 256), 256, 0, stream>>>(W, wT, Kd, Nout);
  };

  // 1) event encoder
  event_encoder_kernel<<<1, 128, 0, stream>>>(ev_feat, etype, eint, ett,
                                              ev_w1, ev_b1, ev_g1, ev_bt1,
                                              ev_w2, ev_b2, ev_g2, ev_bt2, e_buf);
  // 2) x = [emb | e] (bf16)
  build_x256_kernel<<<cdiv((long long)N * 256, 256), 256, 0, stream>>>(node_ids, emb, e_buf,
                                                                       x256b, N * 256);
  // 3) q/k/v/skip GEMMs (WMMA)
  convw(wq, 256, 512);    gemm(x256b, wT, bq,    qbuf, N, 256, 512);
  convw(wk, 256, 512);    gemm(x256b, wT, bk,    kbuf, N, 256, 512);
  convw(wv, 256, 512);    gemm(x256b, wT, bv,    vbuf, N, 256, 512);
  convw(wskip, 256, 128); gemm(x256b, wT, bskip, xcur, N, 256, 128);  // xcur pre-filled w/ skip

  // 4) TransformerConv attention + aggregation (head mean folded: scale 0.25)
  fill_u32_kernel<<<cdiv(N * 4, 256), 256, 0, stream>>>(mEnc, 0u, N * 4);
  fill_f32_kernel<<<cdiv(N * 4, 256), 256, 0, stream>>>(ssum, 0.0f, N * 4);
  attn_logits_kernel<<<cdiv((long long)E * 32, 256), 256, 0, stream>>>(qbuf, kbuf, src, dst,
                                                                       E, logits, mEnc);
  softmax_exp_kernel<<<cdiv((long long)E * 4, 256), 256, 0, stream>>>(logits, dst, E, E, mEnc, ssum);
  aggregate_kernel<<<cdiv((long long)E * 32, 256), 256, 0, stream>>>(vbuf, logits, ssum, src, dst,
                                                                     E, E, 0.25f, xcur);

  // 5) two GAT layers (self loops appended, head mean, relu)
  float* xh = qbuf;  // alias: q no longer needed
  for (int layer = 0; layer < 2; ++layer) {
    f32_to_bf16_kernel<<<cdiv((long long)N * 128, 256), 256, 0, stream>>>(xcur, xb16, N * 128);
    convw(g_w[layer], 128, 512);
    gemm(xb16, wT, nullptr, xh, N, 128, 512);
    gat_coef_kernel<<<cdiv((long long)N * 32, 256), 256, 0, stream>>>(xh, g_as[layer], g_ad[layer],
                                                                      asb, adb, N);
    fill_u32_kernel<<<cdiv(N * 4, 256), 256, 0, stream>>>(mEnc, 0u, N * 4);
    fill_f32_kernel<<<cdiv(N * 4, 256), 256, 0, stream>>>(ssum, 0.0f, N * 4);
    fill_f32_kernel<<<cdiv((long long)N * 128, 256), 256, 0, stream>>>(accb, 0.0f, N * 128);
    gat_logits_kernel<<<cdiv((long long)E2 * 4, 256), 256, 0, stream>>>(asb, adb, src, dst,
                                                                        E, E2, logits, mEnc);
    softmax_exp_kernel<<<cdiv((long long)E2 * 4, 256), 256, 0, stream>>>(logits, dst, E, E2,
                                                                         mEnc, ssum);
    aggregate_kernel<<<cdiv((long long)E2 * 32, 256), 256, 0, stream>>>(xh, logits, ssum, src, dst,
                                                                        E, E2, 0.25f, accb);
    gat_final_kernel<<<cdiv((long long)N * 128, 256), 256, 0, stream>>>(accb, g_b[layer], xcur,
                                                                        N * 128);
  }

  // 6) LSTM cell (h0=c0=0): gates = x@wih + (bih+bhh)
  bias_sum_kernel<<<2, 256, 0, stream>>>(bih, bhh, bs, 512);
  f32_to_bf16_kernel<<<cdiv((long long)N * 128, 256), 256, 0, stream>>>(xcur, xb16, N * 128);
  convw(wih, 128, 512);
  float* gates = qbuf;  // alias
  gemm(xb16, wT, bs, gates, N, 128, 512);
  lstm_final_kernel<<<cdiv((long long)N * 128, 256), 256, 0, stream>>>(gates, (float*)d_out, N);
}
